// GemLiteLinearTriton_29231547416871
// MI455X (gfx1250) — 8-bit matrix operand order repaired and checked by reading, compiled, not run on hardware
//
#include <hip/hip_runtime.h>
#include <cstdint>

typedef int   v8i __attribute__((ext_vector_type(8)));
typedef int   v4i __attribute__((ext_vector_type(4)));
typedef int   v2i __attribute__((ext_vector_type(2)));
typedef float v4f __attribute__((ext_vector_type(4)));

__global__ void __launch_bounds__(256)
quantize_rows_kernel(const float* __restrict__ x,
                     int8_t* __restrict__ xq,
                     float* __restrict__ sx,
                     int M, int K) {
  const int row = blockIdx.x;
  const int tid = threadIdx.x;
  int* dst = (int*)(xq + (size_t)row * K);

  if (row >= M) {
    for (int i = tid; i < (K >> 2); i += 256) dst[i] = 0;
    if (tid == 0) sx[row] = 0.0f;
    return;
  }

  __shared__ float red[256];
  const float* xr = x + (size_t)row * K;

  float m = 0.0f;
  for (int i = tid; i < K; i += 256) m = fmaxf(m, __builtin_fabsf(xr[i]));
  red[tid] = m;
  __syncthreads();
  for (int s = 128; s > 0; s >>= 1) {
    if (tid < s) red[tid] = fmaxf(red[tid], red[tid + s]);
    __syncthreads();
  }
  const float amax   = red[0];
  const float scale  = amax * (1.0f / 127.0f);
  const float rscale = (amax > 0.0f) ? (127.0f / amax) : 0.0f;
  if (tid == 0) sx[row] = scale;

  const v4f* src = (const v4f*)xr;
  for (int i = tid; i < (K >> 2); i += 256) {
    v4f v = src[i];
    unsigned q0 = (unsigned)(int)__builtin_rintf(v.x * rscale) & 0xFFu;
    unsigned q1 = (unsigned)(int)__builtin_rintf(v.y * rscale) & 0xFFu;
    unsigned q2 = (unsigned)(int)__builtin_rintf(v.z * rscale) & 0xFFu;
    unsigned q3 = (unsigned)(int)__builtin_rintf(v.w * rscale) & 0xFFu;
    dst[i] = (int)(q0 | (q1 << 8) | (q2 << 16) | (q3 << 24));
  }
}

__device__ __forceinline__ int pack_w(v4i w) {
  unsigned lo = __builtin_amdgcn_perm((unsigned)w.y, (unsigned)w.x, 0x0c0c0400u);
  unsigned hi = __builtin_amdgcn_perm((unsigned)w.w, (unsigned)w.z, 0x04000c0cu);
  return (int)((lo | hi) ^ 0x80808080u);
}

__device__ __forceinline__ v8i
wmma_step(const int8_t* __restrict__ aptr, const int* __restrict__ bptr,
          int kb, v8i acc) {
  const int8_t* ap = aptr + kb;
  v2i a0 = *(const v2i*)(ap +  0);
  v2i a1 = *(const v2i*)(ap +  8);
  v2i a2 = *(const v2i*)(ap + 32);
  v2i a3 = *(const v2i*)(ap + 40);
  v8i A;
  A[0] = a0.x; A[1] = a0.y; A[2] = a1.x; A[3] = a1.y;
  A[4] = a2.x; A[5] = a2.y; A[6] = a3.x; A[7] = a3.y;

  const v4i* bp0 = (const v4i*)(bptr + kb);
  const v4i* bp1 = (const v4i*)(bptr + kb + 32);
  v8i B;
  B[0] = pack_w(__builtin_nontemporal_load(bp0 + 0));
  B[1] = pack_w(__builtin_nontemporal_load(bp0 + 1));
  B[2] = pack_w(__builtin_nontemporal_load(bp0 + 2));
  B[3] = pack_w(__builtin_nontemporal_load(bp0 + 3));
  B[4] = pack_w(__builtin_nontemporal_load(bp1 + 0));
  B[5] = pack_w(__builtin_nontemporal_load(bp1 + 1));
  B[6] = pack_w(__builtin_nontemporal_load(bp1 + 2));
  B[7] = pack_w(__builtin_nontemporal_load(bp1 + 3));

  return __builtin_amdgcn_wmma_i32_16x16x64_iu8(true, A, true, B, acc,
                                                false, false);
}

__global__ void __launch_bounds__(256)
gemm_wmma_iu8_kernel(const int8_t* __restrict__ xq,
                     const float* __restrict__ sx,
                     const int* __restrict__ Wq,
                     const float* __restrict__ sw,
                     const float* __restrict__ bias,
                     float* __restrict__ out,
                     int M, int K, int N) {
  const int lane   = threadIdx.x & 31;
  const int wave   = threadIdx.x >> 5;
  const int n_base = (blockIdx.x * 8 + wave) * 16;
  if (n_base >= N) return;

  const int half = lane >> 4;
  const int l15  = lane & 15;

  const int8_t* aptr = xq + (size_t)l15 * K + (half << 4);
  const int* bptr = Wq + (size_t)(n_base + l15) * K + (half << 4);

  v8i acc0 = {};
  v8i acc1 = {};
  v8i acc2 = {};
  v8i acc3 = {};
  for (int kb = 0; kb < K; kb += 256) {
    acc0 = wmma_step(aptr, bptr, kb,       acc0);
    acc1 = wmma_step(aptr, bptr, kb +  64, acc1);
    acc2 = wmma_step(aptr, bptr, kb + 128, acc2);
    acc3 = wmma_step(aptr, bptr, kb + 192, acc3);
  }
  v8i acc = (acc0 + acc1) + (acc2 + acc3);

  if (lane < 16) {
    const int n    = n_base + l15;
    const float ws = sw[n];
    const float b  = bias[n];
    #pragma unroll
    for (int i = 0; i < 8; ++i) {
      if (i < M)
        out[(size_t)i * N + n] = (float)acc[i] * sx[i] * ws + b;
    }
  }
}

extern "C" void kernel_launch(void* const* d_in, const int* in_sizes, int n_in,
                              void* d_out, int out_size, void* d_ws, size_t ws_size,
                              hipStream_t stream) {
  const float* x    = (const float*)d_in[0];
  const int*   Wq   = (const int*)d_in[1];
  const float* sw   = (const float*)d_in[2];
  const float* bias = (const float*)d_in[3];
  float*       out  = (float*)d_out;

  const int N = in_sizes[2];
  const int M = out_size / N;
  const int K = in_sizes[0] / M;

  int8_t* xq = (int8_t*)d_ws;
  float*  sx = (float*)((char*)d_ws + (size_t)16 * K);

  quantize_rows_kernel<<<16, 256, 0, stream>>>(x, xq, sx, M, K);

  const int tiles  = (N + 15) / 16;
  const int blocks = (tiles + 7) / 8;
  gemm_wmma_iu8_kernel<<<blocks, 256, 0, stream>>>(xq, sx, Wq, sw, bias, out,
                                                   M, K, N);
}
